// EdgeFrontierPolicy_52793738003057
// MI455X (gfx1250) — matrix operand order repaired and checked by reading, not compiled, not run on hardware
//
#include <hip/hip_runtime.h>
#include <math.h>

typedef __attribute__((ext_vector_type(16))) __bf16 v16bf;
typedef __attribute__((ext_vector_type(8)))  float  v8f;

#define HDIM 128

// ---------------------------------------------------------------- helpers

__device__ __forceinline__ float gelu_f(float x) {
  // erf-based GELU (torch default)
  return 0.5f * x * (1.0f + erff(x * 0.70710678118654752440f));
}

__device__ __forceinline__ v8f wmma_bf16(v16bf a, v16bf b, v8f c) {
  return __builtin_amdgcn_wmma_f32_16x16x32_bf16(
      /*neg_a=*/false, a, /*neg_b=*/false, b,
      /*c_mod=*/(short)0, c, /*reuse_a=*/false, /*reuse_b=*/false);
}

// A fragment: 16x32 bf16 tile from LDS, row-major with leading dim ldk.
// Lane layout (ISA 7.12.2): lane&15 = row M, lane>>4 selects K sub-offset;
// elements 0..7 -> K = koff+0..7, elements 8..15 -> K = koff+16..23.
__device__ __forceinline__ v16bf load_afrag(const __bf16* sA, int ldk, int kt, int lane) {
  int m    = lane & 15;
  int koff = kt * 32 + ((lane >> 4) << 3);
  const __bf16* p = sA + m * ldk + koff;
  v16bf a;
#pragma unroll
  for (int i = 0; i < 8; ++i) { a[i] = p[i]; a[8 + i] = p[16 + i]; }
  return a;
}

// B fragment: packed layout, 512 bf16 per (kt,nt) fragment, 16 contiguous per lane.
__device__ __forceinline__ v16bf load_bfrag(const __bf16* Wp, int kt, int nt, int lane) {
  return *(const v16bf*)(Wp + (((size_t)(kt * 8 + nt)) * 32 + (size_t)lane) * 16);
}

// 16xN (N=128) GEMM: wave w owns N-tiles {w, w+4}.
__device__ __forceinline__ void gemm16(const __bf16* sA, int lda, const __bf16* Wp,
                                       int ktiles, int lane, int wave,
                                       v8f& acc0, v8f& acc1) {
  v8f z = {0.f, 0.f, 0.f, 0.f, 0.f, 0.f, 0.f, 0.f};
  acc0 = z; acc1 = z;
  for (int kt = 0; kt < ktiles; ++kt) {
    v16bf a  = load_afrag(sA, lda, kt, lane);
    v16bf b0 = load_bfrag(Wp, kt, wave, lane);
    v16bf b1 = load_bfrag(Wp, kt, wave + 4, lane);
    acc0 = wmma_bf16(a, b0, acc0);
    acc1 = wmma_bf16(a, b1, acc1);
  }
}

// LayerNorm of a 16-row tile held in LDS (f32, stride ldx), writing OutT with
// stride ldo, zero-padding columns [cols, padcols). 128 threads: 8 per row.
template <typename OutT>
__device__ __forceinline__ void ln_tile(float* sX, int ldx, int cols,
                                        const float* gamma, const float* beta,
                                        OutT* out, int ldo, int padcols,
                                        float* sRed, int tid) {
  __syncthreads();
  int r = tid >> 3, j = tid & 7;
  float s1 = 0.f, s2 = 0.f;
  for (int c = j; c < cols; c += 8) { float x = sX[r * ldx + c]; s1 += x; s2 += x * x; }
  sRed[r * 16 + j] = s1;
  sRed[r * 16 + 8 + j] = s2;
  __syncthreads();
  float S1 = 0.f, S2 = 0.f;
#pragma unroll
  for (int i = 0; i < 8; ++i) { S1 += sRed[r * 16 + i]; S2 += sRed[r * 16 + 8 + i]; }
  float mean = S1 / (float)cols;
  float var  = S2 / (float)cols - mean * mean;
  var = var < 0.f ? 0.f : var;
  float inv = rsqrtf(var + 1e-5f);
  for (int c = j; c < cols; c += 8)
    out[r * ldo + c] = (OutT)((sX[r * ldx + c] - mean) * inv * gamma[c] + beta[c]);
  for (int c = cols + j; c < padcols; c += 8)
    out[r * ldo + c] = (OutT)0.f;
  __syncthreads();
}

__device__ __forceinline__ int lower_bound_i(const int* a, int n, int key) {
  int lo = 0, hi = n;
  while (lo < hi) { int mid = (lo + hi) >> 1; if (a[mid] < key) lo = mid + 1; else hi = mid; }
  return lo;
}

// ---------------------------------------------------------------- kernels

// Pack f32 weight [K x N] (row-major, N=128) into bf16 B-fragments, K padded
// to Kpad (zero fill). Element mapping matches load_bfrag / ISA B layout:
// lane&15 = col-in-tile, k = kt*32 + (lane>>4)*8 + e (e<8), kt*32 + 16 + (lane>>4)*8 + (e-8) (e>=8).
__global__ void pack_w_kernel(const float* __restrict__ W, __bf16* __restrict__ out,
                              int K, int N, int Kpad) {
  int ktiles = Kpad / 32;
  int ntiles = N / 16;
  int total = ktiles * ntiles * 512;
  int idx = blockIdx.x * blockDim.x + threadIdx.x;
  if (idx >= total) return;
  int e = idx & 15;
  int lane = (idx >> 4) & 31;
  int frag = idx >> 9;
  int nt = frag % ntiles;
  int kt = frag / ntiles;
  int n = nt * 16 + (lane & 15);
  int k = kt * 32 + ((lane >> 4) * 8) + (e < 8 ? e : 8 + e);
  float v = (k < K) ? W[(size_t)k * N + n] : 0.f;
  out[idx] = (__bf16)v;
}

// qf = gelu(LN(q) @ qf_W + qf_bias), 16 graphs per block.
__global__ __launch_bounds__(128) void qf_kernel(
    const float* __restrict__ q, const float* __restrict__ g, const float* __restrict__ b,
    const __bf16* __restrict__ Wp, const float* __restrict__ bias,
    float* __restrict__ qf) {
  __shared__ float  sX[16 * HDIM];
  __shared__ __bf16 sA[16 * HDIM];
  __shared__ float  sRed[256];
  int tid = threadIdx.x, lane = tid & 31, wave = tid >> 5;
  int gb = blockIdx.x * 16;
  __builtin_prefetch(Wp, 0, 3);
  for (int idx = tid; idx < 16 * HDIM; idx += 128) {
    int r = idx >> 7, c = idx & 127;
    sX[idx] = q[(size_t)(gb + r) * HDIM + c];
  }
  ln_tile(sX, HDIM, HDIM, g, b, sA, HDIM, HDIM, sRed, tid);
  v8f a0, a1;
  gemm16(sA, HDIM, Wp, 4, lane, wave, a0, a1);
  int m0 = (lane >> 4) << 3;
  int c0 = wave * 16 + (lane & 15);
  int c1 = (wave + 4) * 16 + (lane & 15);
#pragma unroll
  for (int v = 0; v < 8; ++v) {
    int gg = gb + m0 + v;
    qf[(size_t)gg * HDIM + c0] = gelu_f(a0[v] + bias[c0]);
    qf[(size_t)gg * HDIM + c1] = gelu_f(a1[v] + bias[c1]);
  }
}

// edge_repr = gelu(gelu(LN([et|0|frontier]) @ ep_W1 + b1) @ ep_W2 + b2)
__global__ __launch_bounds__(128) void edge_repr_kernel(
    const float* __restrict__ edge_tokens, const float* __restrict__ order_emb,
    const float* __restrict__ qf, const int* __restrict__ edge_batch,
    const unsigned char* __restrict__ sel, const int* __restrict__ sel_order,
    const int* __restrict__ heads, const int* __restrict__ tails,
    const int* __restrict__ cur_tail,
    const float* __restrict__ ep_g, const float* __restrict__ ep_b,
    const __bf16* __restrict__ W1p, const float* __restrict__ b1,
    const __bf16* __restrict__ W2p, const float* __restrict__ b2,
    float* __restrict__ er, int E) {
  __shared__ float  sX[16 * 130];
  __shared__ __bf16 sA[16 * 160];
  __shared__ __bf16 sB[16 * HDIM];
  __shared__ float  sRed[256];
  __shared__ int    rb[16];
  __shared__ int    roi[16];
  __shared__ float  rfr[16];
  int tid = threadIdx.x, lane = tid & 31, wave = tid >> 5;
  int ebase = blockIdx.x * 16;
  __builtin_prefetch(W1p, 0, 3);
  if (tid < 16) {
    int e = ebase + tid;
    int bb = 0, oi = 0; float fr = 0.f;
    if (e < E) {
      bb = edge_batch[e];
      int so = sel_order[e];
      so = so < -1 ? -1 : (so > 10 ? 10 : so);
      oi = so + 1;
      int ct = cur_tail[bb];
      fr = ((sel[e] == 0) && (heads[e] == ct || tails[e] == ct)) ? 1.f : 0.f;
    }
    rb[tid] = bb; roi[tid] = oi; rfr[tid] = fr;
  }
  __syncthreads();
  for (int idx = tid; idx < 16 * 130; idx += 128) {
    int r = idx / 130, c = idx - r * 130;
    int e = ebase + r;
    float v = 0.f;
    if (e < E) {
      if (c < HDIM)
        v = edge_tokens[(size_t)e * HDIM + c] + order_emb[(size_t)roi[r] * HDIM + c] +
            qf[(size_t)rb[r] * HDIM + c];
      else if (c == 129)
        v = rfr[r];
    }
    sX[r * 130 + c] = v;
  }
  ln_tile(sX, 130, 130, ep_g, ep_b, sA, 160, 160, sRed, tid);
  v8f a0, a1;
  gemm16(sA, 160, W1p, 5, lane, wave, a0, a1);
  int m0 = (lane >> 4) << 3;
  int c0 = wave * 16 + (lane & 15);
  int c1 = (wave + 4) * 16 + (lane & 15);
#pragma unroll
  for (int v = 0; v < 8; ++v) {
    sB[(m0 + v) * HDIM + c0] = (__bf16)gelu_f(a0[v] + b1[c0]);
    sB[(m0 + v) * HDIM + c1] = (__bf16)gelu_f(a1[v] + b1[c1]);
  }
  __syncthreads();
  gemm16(sB, HDIM, W2p, 4, lane, wave, a0, a1);
#pragma unroll
  for (int v = 0; v < 8; ++v) {
    int e = ebase + m0 + v;
    if (e < E) {
      er[(size_t)e * HDIM + c0] = gelu_f(a0[v] + b2[c0]);
      er[(size_t)e * HDIM + c1] = gelu_f(a1[v] + b2[c1]);
    }
  }
}

// Segment sum over sorted edge_batch: one block (128 threads) per graph,
// thread t owns column t (coalesced 512B rows).
__global__ __launch_bounds__(128) void segsum_kernel(
    const float* __restrict__ er, const int* __restrict__ edge_batch,
    const unsigned char* __restrict__ sel,
    float* __restrict__ selsum, float* __restrict__ cnt, int E) {
  __shared__ int bounds[2];
  int g = blockIdx.x, t = threadIdx.x;
  if (t == 0) bounds[0] = lower_bound_i(edge_batch, E, g);
  if (t == 1) bounds[1] = lower_bound_i(edge_batch, E, g + 1);
  __syncthreads();
  int s = bounds[0], en = bounds[1];
  float sum = 0.f, c = 0.f;
  for (int e = s; e < en; ++e) {
    if (sel[e]) { sum += er[(size_t)e * HDIM + t]; c += 1.f; }
  }
  selsum[(size_t)g * HDIM + t] = sum;
  if (t == 0) cnt[g] = c;
}

// current_state = LN(sel_sum/cnt + q); stop head on [cs | q].
__global__ __launch_bounds__(128) void graph_head_kernel(
    const float* __restrict__ q, const float* __restrict__ selsum,
    const float* __restrict__ cnt,
    const float* __restrict__ graph_g, const float* __restrict__ graph_b,
    const float* __restrict__ st_g, const float* __restrict__ st_b,
    const __bf16* __restrict__ W1p, const float* __restrict__ b1,
    const float* __restrict__ W2, const float* __restrict__ b2,
    float* __restrict__ cs_out, float* __restrict__ stop_out) {
  __shared__ float  sX[16 * 256];
  __shared__ __bf16 sA[16 * 256];
  __shared__ float  sH[16 * HDIM];
  __shared__ float  sRed[256];
  int tid = threadIdx.x, lane = tid & 31, wave = tid >> 5;
  int gb = blockIdx.x * 16;
  __builtin_prefetch(W1p, 0, 3);
  for (int idx = tid; idx < 16 * HDIM; idx += 128) {
    int r = idx >> 7, c = idx & 127;
    int g = gb + r;
    float cv = cnt[g]; cv = cv < 1.f ? 1.f : cv;
    float qv = q[(size_t)g * HDIM + c];
    sX[r * 256 + c] = selsum[(size_t)g * HDIM + c] / cv + qv;
    sX[r * 256 + HDIM + c] = qv;
  }
  // LN over cols 0..127 in place -> current_state
  ln_tile(sX, 256, HDIM, graph_g, graph_b, sX, 256, HDIM, sRed, tid);
  for (int idx = tid; idx < 16 * HDIM; idx += 128) {
    int r = idx >> 7, c = idx & 127;
    cs_out[(size_t)(gb + r) * HDIM + c] = sX[r * 256 + c];
  }
  ln_tile(sX, 256, 256, st_g, st_b, sA, 256, 256, sRed, tid);
  v8f a0, a1;
  gemm16(sA, 256, W1p, 8, lane, wave, a0, a1);
  int m0 = (lane >> 4) << 3;
  int c0 = wave * 16 + (lane & 15);
  int c1 = (wave + 4) * 16 + (lane & 15);
#pragma unroll
  for (int v = 0; v < 8; ++v) {
    sH[(m0 + v) * HDIM + c0] = gelu_f(a0[v] + b1[c0]);
    sH[(m0 + v) * HDIM + c1] = gelu_f(a1[v] + b1[c1]);
  }
  __syncthreads();
  int r = tid >> 3, j = tid & 7;
  float p = 0.f;
  for (int c = j; c < HDIM; c += 8) p += sH[r * HDIM + c] * W2[c];
  sRed[r * 16 + j] = p;
  __syncthreads();
  if (j == 0) {
    float s = b2[0];
#pragma unroll
    for (int i = 0; i < 8; ++i) s += sRed[r * 16 + i];
    stop_out[gb + r] = s;
  }
}

// edge_logits: next_state LN, concat with gathered current_state, lookahead head.
__global__ __launch_bounds__(128) void edge_logit_kernel(
    const float* __restrict__ q, const float* __restrict__ er,
    const float* __restrict__ selsum, const float* __restrict__ cnt,
    const float* __restrict__ cs, const int* __restrict__ edge_batch,
    const unsigned char* __restrict__ sel, const int* __restrict__ heads,
    const int* __restrict__ tails, const int* __restrict__ cur_tail,
    const float* __restrict__ graph_g, const float* __restrict__ graph_b,
    const float* __restrict__ lh_g, const float* __restrict__ lh_b,
    const __bf16* __restrict__ W1p, const float* __restrict__ b1,
    const float* __restrict__ W2, const float* __restrict__ b2,
    float* __restrict__ logits, int E) {
  __shared__ float  sX[16 * 256];
  __shared__ __bf16 sA[16 * 256];
  __shared__ float  sH[16 * HDIM];
  __shared__ float  sRed[256];
  __shared__ int    rb[16];
  __shared__ float  rfr[16];
  int tid = threadIdx.x, lane = tid & 31, wave = tid >> 5;
  int ebase = blockIdx.x * 16;
  __builtin_prefetch(W1p, 0, 3);
  if (tid < 16) {
    int e = ebase + tid;
    int bb = 0; float fr = 0.f;
    if (e < E) {
      bb = edge_batch[e];
      int ct = cur_tail[bb];
      fr = ((sel[e] == 0) && (heads[e] == ct || tails[e] == ct)) ? 1.f : 0.f;
    }
    rb[tid] = bb; rfr[tid] = fr;
  }
  __syncthreads();
  for (int idx = tid; idx < 16 * HDIM; idx += 128) {
    int r = idx >> 7, c = idx & 127;
    int e = ebase + r;
    int bb = rb[r];
    float npre = 0.f, csv = 0.f;
    if (e < E) {
      float nc = cnt[bb] + 1.f;
      npre = (selsum[(size_t)bb * HDIM + c] + er[(size_t)e * HDIM + c]) / nc +
             q[(size_t)bb * HDIM + c];
      csv = cs[(size_t)bb * HDIM + c];
    }
    sX[r * 256 + c] = csv;
    sX[r * 256 + HDIM + c] = npre;
  }
  // next_state = LN(npre) in place (cols 128..255)
  ln_tile(sX + HDIM, 256, HDIM, graph_g, graph_b, sX + HDIM, 256, HDIM, sRed, tid);
  ln_tile(sX, 256, 256, lh_g, lh_b, sA, 256, 256, sRed, tid);
  v8f a0, a1;
  gemm16(sA, 256, W1p, 8, lane, wave, a0, a1);
  int m0 = (lane >> 4) << 3;
  int c0 = wave * 16 + (lane & 15);
  int c1 = (wave + 4) * 16 + (lane & 15);
#pragma unroll
  for (int v = 0; v < 8; ++v) {
    sH[(m0 + v) * HDIM + c0] = gelu_f(a0[v] + b1[c0]);
    sH[(m0 + v) * HDIM + c1] = gelu_f(a1[v] + b1[c1]);
  }
  __syncthreads();
  int r = tid >> 3, j = tid & 7;
  float p = 0.f;
  for (int c = j; c < HDIM; c += 8) p += sH[r * HDIM + c] * W2[c];
  sRed[r * 16 + j] = p;
  __syncthreads();
  if (j == 0) {
    int e = ebase + r;
    if (e < E) {
      float s = b2[0];
#pragma unroll
      for (int i = 0; i < 8; ++i) s += sRed[r * 16 + i];
      logits[e] = s + 0.5f * rfr[r];
    }
  }
}

// ---------------------------------------------------------------- launch

extern "C" void kernel_launch(void* const* d_in, const int* in_sizes, int n_in,
                              void* d_out, int out_size, void* d_ws, size_t ws_size,
                              hipStream_t stream) {
  (void)n_in; (void)out_size; (void)ws_size;
  const int E = in_sizes[0] / HDIM;   // 500000
  const int G = in_sizes[1] / HDIM;   // 2048

  const float* edge_tokens = (const float*)d_in[0];
  const float* question    = (const float*)d_in[1];
  const int*   edge_batch  = (const int*)d_in[2];
  const unsigned char* selm = (const unsigned char*)d_in[3];  // bool -> u8
  const int*   sel_order   = (const int*)d_in[4];
  const int*   heads       = (const int*)d_in[5];
  const int*   tails       = (const int*)d_in[6];
  const int*   cur_tail    = (const int*)d_in[7];
  // params in dict insertion order
  const float* graph_g = (const float*)d_in[8];
  const float* graph_b = (const float*)d_in[9];
  const float* qf_g    = (const float*)d_in[10];
  const float* qf_b    = (const float*)d_in[11];
  const float* qf_W    = (const float*)d_in[12];
  const float* qf_bias = (const float*)d_in[13];
  const float* ep_g    = (const float*)d_in[14];
  const float* ep_b    = (const float*)d_in[15];
  const float* ep_W1   = (const float*)d_in[16];
  const float* ep_b1   = (const float*)d_in[17];
  const float* ep_W2   = (const float*)d_in[18];
  const float* ep_b2   = (const float*)d_in[19];
  const float* lh_g    = (const float*)d_in[20];
  const float* lh_b    = (const float*)d_in[21];
  const float* lh_W1   = (const float*)d_in[22];
  const float* lh_b1   = (const float*)d_in[23];
  const float* lh_W2   = (const float*)d_in[24];
  const float* lh_b2   = (const float*)d_in[25];
  const float* st_g    = (const float*)d_in[26];
  const float* st_b    = (const float*)d_in[27];
  const float* st_W1   = (const float*)d_in[28];
  const float* st_b1   = (const float*)d_in[29];
  const float* st_W2   = (const float*)d_in[30];
  const float* st_b2   = (const float*)d_in[31];
  const float* order_emb = (const float*)d_in[32];

  float* out        = (float*)d_out;
  float* logits_out = out;                       // [E]
  float* stop_out   = out + E;                   // [G]
  float* cs_out     = out + E + G;               // [G*H]

  // workspace layout
  float* qf     = (float*)d_ws;
  float* selsum = qf + (size_t)G * HDIM;
  float* cnt    = selsum + (size_t)G * HDIM;
  float* er     = cnt + G;
  __bf16* wp_qf  = (__bf16*)(er + (size_t)E * HDIM);
  __bf16* wp_ep1 = wp_qf  + (size_t)4 * 8 * 512;   // K=128 -> 4 ktiles
  __bf16* wp_ep2 = wp_ep1 + (size_t)5 * 8 * 512;   // Kpad=160 -> 5 ktiles
  __bf16* wp_lh1 = wp_ep2 + (size_t)4 * 8 * 512;
  __bf16* wp_st1 = wp_lh1 + (size_t)8 * 8 * 512;   // K=256 -> 8 ktiles

  // 1) pack weights to bf16 B-fragments
  {
    int t;
    t = 4 * 8 * 512; pack_w_kernel<<<(t + 255) / 256, 256, 0, stream>>>(qf_W,  wp_qf,  128, 128, 128);
    t = 5 * 8 * 512; pack_w_kernel<<<(t + 255) / 256, 256, 0, stream>>>(ep_W1, wp_ep1, 130, 128, 160);
    t = 4 * 8 * 512; pack_w_kernel<<<(t + 255) / 256, 256, 0, stream>>>(ep_W2, wp_ep2, 128, 128, 128);
    t = 8 * 8 * 512; pack_w_kernel<<<(t + 255) / 256, 256, 0, stream>>>(lh_W1, wp_lh1, 256, 128, 256);
    t = 8 * 8 * 512; pack_w_kernel<<<(t + 255) / 256, 256, 0, stream>>>(st_W1, wp_st1, 256, 128, 256);
  }

  // 2) qf per graph
  qf_kernel<<<G / 16, 128, 0, stream>>>(question, qf_g, qf_b, wp_qf, qf_bias, qf);

  // 3) edge_repr
  int eblocks = (E + 15) / 16;
  edge_repr_kernel<<<eblocks, 128, 0, stream>>>(
      edge_tokens, order_emb, qf, edge_batch, selm, sel_order, heads, tails, cur_tail,
      ep_g, ep_b, wp_ep1, ep_b1, wp_ep2, ep_b2, er, E);

  // 4) segment sums
  segsum_kernel<<<G, 128, 0, stream>>>(er, edge_batch, selm, selsum, cnt, E);

  // 5) current_state + stop head
  graph_head_kernel<<<G / 16, 128, 0, stream>>>(
      question, selsum, cnt, graph_g, graph_b, st_g, st_b,
      wp_st1, st_b1, st_W2, st_b2, cs_out, stop_out);

  // 6) edge logits
  edge_logit_kernel<<<eblocks, 128, 0, stream>>>(
      question, er, selsum, cnt, cs_out, edge_batch, selm, heads, tails, cur_tail,
      graph_g, graph_b, lh_g, lh_b, wp_lh1, lh_b1, lh_W2, lh_b2, logits_out, E);
}
